// SNN_31825707664217
// MI455X (gfx1250) — matrix operand order repaired and checked by reading, not compiled, not run on hardware
//
#include <hip/hip_runtime.h>

typedef __bf16 v16bf __attribute__((ext_vector_type(16)));
typedef __bf16 v8bf  __attribute__((ext_vector_type(8)));
typedef float  v8f   __attribute__((ext_vector_type(8)));

// global (addrspace 1) pointer types: keep weight loads as global_load_b128
// (not flat_load_b128) even after laundering the base pointer.
typedef __attribute__((address_space(1))) const __bf16 gbf16;
typedef __attribute__((address_space(1))) const v8bf  gv8bf;

#define B_SZ   128
#define T_STEPS 200
#define IN_D   700
#define IN_P   704
#define H_D    512
#define O_D    20

// bf16 workspace layout (element offsets), all 16B-aligned, rows padded
#define OFF_W1X  0          // [512][704]
#define OFF_W1R  360448     // [512][512]
#define OFF_W1TM 622592     // [512][1024]
#define OFF_W1TA 1146880    // [512][1024]
#define OFF_W2X  1671168    // [512][512]
#define OFF_W2R  1933312    // [512][512]
#define OFF_W2TM 2195456    // [512][1024]
#define OFF_W2TA 2719744    // [512][1024]
#define OFF_W3X  3244032    // [32][512] (rows >=20 zero)
#define OFF_W3TM 3260416    // [32][64]  (rows>=20, cols>=40 zero)
#define WS_ELEMS 3262464

// d_out layout (floats)
#define OUT_MEM1 512000
#define OUT_SPK1 577536
#define OUT_B1   643072
#define OUT_MEM2 708608
#define OUT_SPK2 774144
#define OUT_B2   839680
#define OUT_MEM3 905216

// ---- WMMA helpers (CDNA5 16x16x32 bf16, wave32 layouts per ISA 7.12.2) ----
__device__ __forceinline__ v16bf ldsA(const __bf16* base, int ldk, int k0, int lane) {
  // A 16x32: lane<16 -> K {0..7, 16..23}; lane>=16 -> K {8..15, 24..31}; M = lane&15
  int m  = lane & 15;
  int kh = (lane & 16) ? 8 : 0;
  const __bf16* p = base + m * ldk + k0 + kh;
  union { v16bf v; v8bf h[2]; } u;
  u.h[0] = *(const v8bf*)(p);
  u.h[1] = *(const v8bf*)(p + 16);
  return u.v;
}
__device__ __forceinline__ v16bf glbB(const gbf16* W, int ldk, int n0, int k0, int lane) {
  // B 32x16 from row-major W[N][K]: lane<16 -> col n0+lane, K {0..7, 16..23}; lane>=16 -> K {8..15, 24..31}
  int n  = n0 + (lane & 15);
  int kb = k0 + ((lane & 16) ? 8 : 0);
  const gbf16* p = W + (size_t)n * ldk + kb;
  union { v16bf v; v8bf h[2]; } u;
  u.h[0] = *(const gv8bf*)(p);
  u.h[1] = *(const gv8bf*)(p + 16);
  return u.v;
}
#define WMMA(acc, a, b) (acc) = __builtin_amdgcn_wmma_f32_16x16x32_bf16(false, (a), false, (b), (short)0, (acc), false, false)

__device__ __forceinline__ float sigm(float z) { return 1.0f / (1.0f + __expf(-z)); }

// ---- weight f32 -> bf16 prep (padded layouts) ----
__global__ void snn_prep(const float* __restrict__ W1x, const float* __restrict__ W1r,
                         const float* __restrict__ W1tm, const float* __restrict__ W1ta,
                         const float* __restrict__ W2x, const float* __restrict__ W2r,
                         const float* __restrict__ W2tm, const float* __restrict__ W2ta,
                         const float* __restrict__ W3x, const float* __restrict__ W3tm,
                         __bf16* __restrict__ ws) {
  for (int idx = blockIdx.x * blockDim.x + threadIdx.x; idx < WS_ELEMS;
       idx += gridDim.x * blockDim.x) {
    float v;
    if (idx < OFF_W1R)       { int i = idx;          int n = i / IN_P, k = i % IN_P; v = (k < IN_D) ? W1x[n * IN_D + k] : 0.f; }
    else if (idx < OFF_W1TM)   v = W1r[idx - OFF_W1R];
    else if (idx < OFF_W1TA)   v = W1tm[idx - OFF_W1TM];
    else if (idx < OFF_W2X)    v = W1ta[idx - OFF_W1TA];
    else if (idx < OFF_W2R)    v = W2x[idx - OFF_W2X];
    else if (idx < OFF_W2TM)   v = W2r[idx - OFF_W2R];
    else if (idx < OFF_W2TA)   v = W2tm[idx - OFF_W2TM];
    else if (idx < OFF_W3X)    v = W2ta[idx - OFF_W2TA];
    else if (idx < OFF_W3TM) { int i = idx - OFF_W3X;  int n = i >> 9, k = i & 511; v = (n < O_D) ? W3x[n * H_D + k] : 0.f; }
    else                     { int i = idx - OFF_W3TM; int n = i >> 6, k = i & 63;  v = (n < O_D && k < 40) ? W3tm[n * 40 + k] : 0.f; }
    ws[idx] = (__bf16)v;
  }
}

// ---- persistent per-batch-block scan kernel: 8 WGs x 512 threads (16 waves) ----
__global__ __launch_bounds__(512, 1)
void snn_scan(const float* __restrict__ xin,
              const float* __restrict__ m1_0, const float* __restrict__ s1_0, const float* __restrict__ b1_0,
              const float* __restrict__ m2_0, const float* __restrict__ s2_0, const float* __restrict__ b2_0,
              const float* __restrict__ m3_0,
              const float* __restrict__ bb1x, const float* __restrict__ bb1r,
              const float* __restrict__ bb1tm, const float* __restrict__ bb1ta,
              const float* __restrict__ bb2x, const float* __restrict__ bb2r,
              const float* __restrict__ bb2tm, const float* __restrict__ bb2ta,
              const float* __restrict__ bb3x, const float* __restrict__ bb3tm,
              const float* __restrict__ pa1m, const float* __restrict__ pa1a,
              const float* __restrict__ pa2m, const float* __restrict__ pa2a,
              const float* __restrict__ pa3,
              const __bf16* __restrict__ ws,
              float* __restrict__ out) {
  __shared__ __attribute__((aligned(16))) __bf16 x_bf[16 * IN_P];
  __shared__ __attribute__((aligned(16))) __bf16 spkA[16 * H_D];
  __shared__ __attribute__((aligned(16))) __bf16 spkB[16 * H_D];
  __shared__ __attribute__((aligned(16))) __bf16 spk2s[16 * H_D];
  __shared__ __attribute__((aligned(16))) __bf16 dense_bf[16 * H_D];
  __shared__ __attribute__((aligned(16))) __bf16 mem1_bf[16 * H_D];
  __shared__ __attribute__((aligned(16))) __bf16 b1_bf[16 * H_D];
  __shared__ __attribute__((aligned(16))) __bf16 mem2_bf[16 * H_D];
  __shared__ __attribute__((aligned(16))) __bf16 b2_bf[16 * H_D];
  __shared__ __attribute__((aligned(16))) __bf16 l3cat[16 * 64];
  __shared__ __attribute__((aligned(16))) float  dense3f[16 * 32];
  __shared__ __attribute__((aligned(16))) float  mem3s[16 * O_D];

  const int tid  = threadIdx.x;
  const int lane = tid & 31;
  const int wv   = tid >> 5;          // 0..15
  const int n0w  = wv * 32;           // this wave owns N columns [n0w, n0w+32)
  const int r0   = blockIdx.x * 16;   // batch rows [r0, r0+16)
  const int mrow = (lane & 16) ? 8 : 0;

  const float a1m = pa1m[0], a1a = pa1a[0], a2m = pa2m[0], a2a = pa2a[0], a3 = pa3[0];

  // persistent f32 state in registers (2 column tiles x 8 rows each)
  v8f mem1r[2], b1r[2], mem2r[2], b2r[2];

  // ---- init: register state + bf16 mirrors + LDS spike/mem3 buffers ----
  for (int t2 = 0; t2 < 2; ++t2) {
    int ncol = n0w + 16 * t2 + (lane & 15);
    #pragma unroll
    for (int i = 0; i < 8; ++i) {
      int m = mrow + i;
      size_t gi = (size_t)(r0 + m) * H_D + ncol;
      float v1 = m1_0[gi], vb1 = b1_0[gi], v2 = m2_0[gi], vb2 = b2_0[gi];
      mem1r[t2][i] = v1; b1r[t2][i] = vb1; mem2r[t2][i] = v2; b2r[t2][i] = vb2;
      mem1_bf[m * H_D + ncol] = (__bf16)v1;  b1_bf[m * H_D + ncol] = (__bf16)vb1;
      mem2_bf[m * H_D + ncol] = (__bf16)v2;  b2_bf[m * H_D + ncol] = (__bf16)vb2;
    }
  }
  for (int idx = tid; idx < 16 * H_D; idx += 512) {
    int m = idx >> 9, n = idx & 511;
    spkA[idx]  = (__bf16)s1_0[(size_t)(r0 + m) * H_D + n];
    spk2s[idx] = (__bf16)s2_0[(size_t)(r0 + m) * H_D + n];
    spkB[idx]  = (__bf16)0.f;
  }
  for (int idx = tid; idx < 16 * O_D; idx += 512) {
    int m = idx / O_D, j = idx % O_D;
    mem3s[idx] = m3_0[(size_t)(r0 + m) * O_D + j];
  }
  if (tid < 64) { int m = tid >> 2, k = IN_D + (tid & 3); x_bf[m * IN_P + k] = (__bf16)0.f; }
  if (tid < 384) { int m = tid / 24, j = 40 + tid % 24; l3cat[m * 64 + j] = (__bf16)0.f; }
  __syncthreads();

  __bf16* spk_cur = spkA;   // spk1(t)
  __bf16* spk_nxt = spkB;   // spk1(t+1)

  // Laundered *global* weight base: prevents LLVM from proving weight loads
  // loop-invariant across the 200-step scan (which caused B-tile spills to
  // scratch), while staying in addrspace(1) so loads lower to global_load_b128
  // (LOADcnt only) instead of flat_load_b128 (LOADcnt+DScnt, LDS contention).
  const gbf16* wsg = (const gbf16*)ws;

  for (int t = 0; t < T_STEPS; ++t) {
    asm volatile("" : "+v"(wsg));
    const gbf16* wsl = wsg;

    // B0: stage x_t (f32 -> bf16) and prefetch next timestep
    for (int idx = tid; idx < 16 * IN_D; idx += 512) {
      int m = idx / IN_D, k = idx - m * IN_D;
      const float* src = &xin[(size_t)(r0 + m) * (T_STEPS * IN_D) + (size_t)t * IN_D + k];
      x_bf[m * IN_P + k] = (__bf16)(*src);
      if (t + 1 < T_STEPS) __builtin_prefetch(src + IN_D, 0, 1);
    }
    __syncthreads();

    v8f dxf[2];  // dense_x / dense_x2 f32 kept in registers

    // ---- S1: dense_x = x@W1x.T + spk1@W1r.T + biases ----
    {
      v8f acc[2] = {{}, {}};
      #pragma unroll 2
      for (int k0 = 0; k0 < IN_P; k0 += 32) {
        v16bf a = ldsA(x_bf, IN_P, k0, lane);
        WMMA(acc[0], a, glbB(wsl + OFF_W1X, IN_P, n0w, k0, lane));
        WMMA(acc[1], a, glbB(wsl + OFF_W1X, IN_P, n0w + 16, k0, lane));
      }
      #pragma unroll 2
      for (int k0 = 0; k0 < H_D; k0 += 32) {
        v16bf a = ldsA(spk_cur, H_D, k0, lane);
        WMMA(acc[0], a, glbB(wsl + OFF_W1R, H_D, n0w, k0, lane));
        WMMA(acc[1], a, glbB(wsl + OFF_W1R, H_D, n0w + 16, k0, lane));
      }
      for (int t2 = 0; t2 < 2; ++t2) {
        int ncol = n0w + 16 * t2 + (lane & 15);
        float bias = bb1x[ncol] + bb1r[ncol];
        #pragma unroll
        for (int i = 0; i < 8; ++i) {
          float d = acc[t2][i] + bias;
          dxf[t2][i] = d;
          dense_bf[(mrow + i) * H_D + ncol] = (__bf16)d;
        }
      }
    }
    __syncthreads();

    // ---- S2a: tauM1/tauA1 GEMMs over K=1024 (concat) ----
    v8f aM[2] = {{}, {}}, aA[2] = {{}, {}};
    #pragma unroll 2
    for (int k0 = 0; k0 < H_D; k0 += 32) {
      v16bf a = ldsA(dense_bf, H_D, k0, lane);
      WMMA(aM[0], a, glbB(wsl + OFF_W1TM, 1024, n0w, k0, lane));
      WMMA(aM[1], a, glbB(wsl + OFF_W1TM, 1024, n0w + 16, k0, lane));
      WMMA(aA[0], a, glbB(wsl + OFF_W1TA, 1024, n0w, k0, lane));
      WMMA(aA[1], a, glbB(wsl + OFF_W1TA, 1024, n0w + 16, k0, lane));
    }
    #pragma unroll 2
    for (int k0 = 0; k0 < H_D; k0 += 32) {
      v16bf am = ldsA(mem1_bf, H_D, k0, lane);
      WMMA(aM[0], am, glbB(wsl + OFF_W1TM, 1024, n0w, 512 + k0, lane));
      WMMA(aM[1], am, glbB(wsl + OFF_W1TM, 1024, n0w + 16, 512 + k0, lane));
      v16bf ab = ldsA(b1_bf, H_D, k0, lane);
      WMMA(aA[0], ab, glbB(wsl + OFF_W1TA, 1024, n0w, 512 + k0, lane));
      WMMA(aA[1], ab, glbB(wsl + OFF_W1TA, 1024, n0w + 16, 512 + k0, lane));
    }
    __syncthreads();

    // ---- S2b: layer-1 elementwise (mem1/b1 in regs, spike out) ----
    for (int t2 = 0; t2 < 2; ++t2) {
      int ncol = n0w + 16 * t2 + (lane & 15);
      float btm = bb1tm[ncol], bta = bb1ta[ncol];
      #pragma unroll
      for (int i = 0; i < 8; ++i) {
        int m = mrow + i;
        float tauM = sigm(a1m * (aM[t2][i] + btm));
        float tauA = sigm(a1a * (aA[t2][i] + bta));
        float spO  = (float)spk_cur[m * H_D + ncol];
        float bN   = tauA * b1r[t2][i] + (1.f - tauA) * spO;
        float Bth  = 0.1f + 1.8f * bN;
        float mO   = mem1r[t2][i];
        float mN   = mO + (dxf[t2][i] - mO) * tauM;
        float sN   = (mN - Bth) > 0.f ? 1.f : 0.f;
        mN = (1.f - sN) * mN;
        mem1r[t2][i] = mN; b1r[t2][i] = bN;
        spk_nxt[m * H_D + ncol] = (__bf16)sN;
        mem1_bf[m * H_D + ncol] = (__bf16)mN;
        b1_bf[m * H_D + ncol]   = (__bf16)bN;
      }
    }
    __syncthreads();

    // ---- S3: dense_x2 = spk1n@W2x.T + spk1@W2r.T + biases ----
    {
      v8f acc[2] = {{}, {}};
      #pragma unroll 2
      for (int k0 = 0; k0 < H_D; k0 += 32) {
        v16bf an = ldsA(spk_nxt, H_D, k0, lane);
        WMMA(acc[0], an, glbB(wsl + OFF_W2X, H_D, n0w, k0, lane));
        WMMA(acc[1], an, glbB(wsl + OFF_W2X, H_D, n0w + 16, k0, lane));
        v16bf ao = ldsA(spk_cur, H_D, k0, lane);
        WMMA(acc[0], ao, glbB(wsl + OFF_W2R, H_D, n0w, k0, lane));
        WMMA(acc[1], ao, glbB(wsl + OFF_W2R, H_D, n0w + 16, k0, lane));
      }
      for (int t2 = 0; t2 < 2; ++t2) {
        int ncol = n0w + 16 * t2 + (lane & 15);
        float bias = bb2x[ncol] + bb2r[ncol];
        #pragma unroll
        for (int i = 0; i < 8; ++i) {
          float d = acc[t2][i] + bias;
          dxf[t2][i] = d;
          dense_bf[(mrow + i) * H_D + ncol] = (__bf16)d;
        }
      }
    }
    __syncthreads();

    // ---- S4a: tauM2/tauA2 GEMMs ----
    v8f cM[2] = {{}, {}}, cA[2] = {{}, {}};
    #pragma unroll 2
    for (int k0 = 0; k0 < H_D; k0 += 32) {
      v16bf a = ldsA(dense_bf, H_D, k0, lane);
      WMMA(cM[0], a, glbB(wsl + OFF_W2TM, 1024, n0w, k0, lane));
      WMMA(cM[1], a, glbB(wsl + OFF_W2TM, 1024, n0w + 16, k0, lane));
      WMMA(cA[0], a, glbB(wsl + OFF_W2TA, 1024, n0w, k0, lane));
      WMMA(cA[1], a, glbB(wsl + OFF_W2TA, 1024, n0w + 16, k0, lane));
    }
    #pragma unroll 2
    for (int k0 = 0; k0 < H_D; k0 += 32) {
      v16bf am = ldsA(mem2_bf, H_D, k0, lane);
      WMMA(cM[0], am, glbB(wsl + OFF_W2TM, 1024, n0w, 512 + k0, lane));
      WMMA(cM[1], am, glbB(wsl + OFF_W2TM, 1024, n0w + 16, 512 + k0, lane));
      v16bf ab = ldsA(b2_bf, H_D, k0, lane);
      WMMA(cA[0], ab, glbB(wsl + OFF_W2TA, 1024, n0w, 512 + k0, lane));
      WMMA(cA[1], ab, glbB(wsl + OFF_W2TA, 1024, n0w + 16, 512 + k0, lane));
    }
    __syncthreads();

    // ---- S4b: layer-2 elementwise ----
    for (int t2 = 0; t2 < 2; ++t2) {
      int ncol = n0w + 16 * t2 + (lane & 15);
      float btm = bb2tm[ncol], bta = bb2ta[ncol];
      #pragma unroll
      for (int i = 0; i < 8; ++i) {
        int m = mrow + i;
        float tauM = sigm(a2m * (cM[t2][i] + btm));
        float tauA = sigm(a2a * (cA[t2][i] + bta));
        float spO  = (float)spk2s[m * H_D + ncol];
        float bN   = tauA * b2r[t2][i] + (1.f - tauA) * spO;
        float Bth  = 0.1f + 1.8f * bN;
        float mO   = mem2r[t2][i];
        float mN   = mO + (dxf[t2][i] - mO) * tauM;
        float sN   = (mN - Bth) > 0.f ? 1.f : 0.f;
        mN = (1.f - sN) * mN;
        mem2r[t2][i] = mN; b2r[t2][i] = bN;
        spk2s[m * H_D + ncol]   = (__bf16)sN;
        mem2_bf[m * H_D + ncol] = (__bf16)mN;
        b2_bf[m * H_D + ncol]   = (__bf16)bN;
      }
    }
    __syncthreads();

    // ---- S5: dense3 = spk2n@W3x.T (waves 0,1) ; mem3 -> concat cols (wave 2) ----
    if (wv < 2) {
      int n03 = 16 * wv;
      v8f acc = {};
      #pragma unroll 2
      for (int k0 = 0; k0 < H_D; k0 += 32) {
        v16bf a = ldsA(spk2s, H_D, k0, lane);
        WMMA(acc, a, glbB(wsl + OFF_W3X, H_D, n03, k0, lane));
      }
      int ncol = n03 + (lane & 15);
      float bias = (ncol < O_D) ? bb3x[ncol] : 0.f;
      #pragma unroll
      for (int i = 0; i < 8; ++i) {
        int m = mrow + i;
        float d = acc[i] + bias;
        dense3f[m * 32 + ncol] = d;
        if (ncol < O_D) l3cat[m * 64 + ncol] = (__bf16)d;
      }
    } else if (wv == 2) {
      for (int idx = lane; idx < 16 * O_D; idx += 32) {
        int m = idx / O_D, j = idx % O_D;
        l3cat[m * 64 + O_D + j] = (__bf16)mem3s[m * O_D + j];
      }
    }
    __syncthreads();

    // ---- S6: tauM3 + mem3 update (waves 0,1) ----
    if (wv < 2) {
      int n03 = 16 * wv;
      v8f acc = {};
      { v16bf a = ldsA(l3cat, 64, 0, lane);  WMMA(acc, a, glbB(wsl + OFF_W3TM, 64, n03, 0, lane)); }
      { v16bf a = ldsA(l3cat, 64, 32, lane); WMMA(acc, a, glbB(wsl + OFF_W3TM, 64, n03, 32, lane)); }
      int ncol = n03 + (lane & 15);
      if (ncol < O_D) {
        float bias = bb3tm[ncol];
        #pragma unroll
        for (int i = 0; i < 8; ++i) {
          int m = mrow + i;
          float tau = sigm(a3 * (acc[i] + bias));
          mem3s[m * O_D + ncol] = (1.f - tau) * mem3s[m * O_D + ncol] + dense3f[m * 32 + ncol] * tau;
        }
      }
    }
    __syncthreads();

    // ---- S7: log_softmax + output store (wave 0, one row per lane) ----
    if (wv == 0 && lane < 16) {
      int m = lane;
      float mx = -3.402823466e+38f;
      for (int j = 0; j < O_D; ++j) mx = fmaxf(mx, mem3s[m * O_D + j]);
      float s = 0.f;
      for (int j = 0; j < O_D; ++j) s += __expf(mem3s[m * O_D + j] - mx);
      float ls = __logf(s);
      size_t base = (size_t)t * (B_SZ * O_D) + (size_t)(r0 + m) * O_D;
      for (int j = 0; j < O_D; ++j) out[base + j] = mem3s[m * O_D + j] - mx - ls;
    }
    { __bf16* tmp = spk_cur; spk_cur = spk_nxt; spk_nxt = tmp; }
    __syncthreads();
  }

  // ---- final state writes ----
  for (int t2 = 0; t2 < 2; ++t2) {
    int ncol = n0w + 16 * t2 + (lane & 15);
    #pragma unroll
    for (int i = 0; i < 8; ++i) {
      int m = mrow + i;
      size_t gi = (size_t)(r0 + m) * H_D + ncol;
      out[OUT_MEM1 + gi] = mem1r[t2][i];
      out[OUT_B1   + gi] = b1r[t2][i];
      out[OUT_MEM2 + gi] = mem2r[t2][i];
      out[OUT_B2   + gi] = b2r[t2][i];
    }
  }
  for (int idx = tid; idx < 16 * H_D; idx += 512) {
    int m = idx >> 9, n = idx & 511;
    size_t gi = (size_t)(r0 + m) * H_D + n;
    out[OUT_SPK1 + gi] = (float)spk_cur[idx];
    out[OUT_SPK2 + gi] = (float)spk2s[idx];
  }
  for (int idx = tid; idx < 16 * O_D; idx += 512) {
    int m = idx / O_D, j = idx % O_D;
    out[OUT_MEM3 + (size_t)(r0 + m) * O_D + j] = mem3s[idx];
  }
}

extern "C" void kernel_launch(void* const* d_in, const int* in_sizes, int n_in,
                              void* d_out, int out_size, void* d_ws, size_t ws_size,
                              hipStream_t stream) {
  const float* xin  = (const float*)d_in[0];
  const float* m1_0 = (const float*)d_in[1];
  const float* s1_0 = (const float*)d_in[2];
  const float* b1_0 = (const float*)d_in[3];
  const float* m2_0 = (const float*)d_in[4];
  const float* s2_0 = (const float*)d_in[5];
  const float* b2_0 = (const float*)d_in[6];
  const float* m3_0 = (const float*)d_in[7];
  const float* W1x  = (const float*)d_in[8];   const float* bb1x = (const float*)d_in[9];
  const float* W1r  = (const float*)d_in[10];  const float* bb1r = (const float*)d_in[11];
  const float* W1tm = (const float*)d_in[12];  const float* bb1tm = (const float*)d_in[13];
  const float* W1ta = (const float*)d_in[14];  const float* bb1ta = (const float*)d_in[15];
  const float* W2x  = (const float*)d_in[16];  const float* bb2x = (const float*)d_in[17];
  const float* W2r  = (const float*)d_in[18];  const float* bb2r = (const float*)d_in[19];
  const float* W2tm = (const float*)d_in[20];  const float* bb2tm = (const float*)d_in[21];
  const float* W2ta = (const float*)d_in[22];  const float* bb2ta = (const float*)d_in[23];
  const float* W3x  = (const float*)d_in[24];  const float* bb3x = (const float*)d_in[25];
  const float* W3tm = (const float*)d_in[26];  const float* bb3tm = (const float*)d_in[27];
  const float* a1m  = (const float*)d_in[28];
  const float* a1a  = (const float*)d_in[29];
  const float* a2m  = (const float*)d_in[30];
  const float* a2a  = (const float*)d_in[31];
  const float* a3   = (const float*)d_in[32];
  __bf16* ws  = (__bf16*)d_ws;
  float*  out = (float*)d_out;

  snn_prep<<<4096, 256, 0, stream>>>(W1x, W1r, W1tm, W1ta, W2x, W2r, W2tm, W2ta, W3x, W3tm, ws);
  snn_scan<<<B_SZ / 16, 512, 0, stream>>>(xin, m1_0, s1_0, b1_0, m2_0, s2_0, b2_0, m3_0,
                                          bb1x, bb1r, bb1tm, bb1ta,
                                          bb2x, bb2r, bb2tm, bb2ta,
                                          bb3x, bb3tm,
                                          a1m, a1a, a2m, a2a, a3,
                                          ws, out);
}
